// CausalSelfAttention_42348377538805
// MI455X (gfx1250) — compile-verified, matrix operand order checked by reading, not run on hardware
//
#include <hip/hip_runtime.h>

// ---------------------------------------------------------------------------
// CDNA5 (gfx1250) causal self-attention: bf16 WMMA everywhere, fp32 accum.
// B=2, T=2048, C=1024, H=16, D=64.
// GEMM wave tile 32Mx64N (8 wmma / 12 loads per k-step); flash attention with
// S^T/O^T formulation (softmax fully in-lane, P^T lands in PV B-fragment).
// ---------------------------------------------------------------------------

typedef __bf16  v16bf __attribute__((ext_vector_type(16)));
typedef float   v8f   __attribute__((ext_vector_type(8)));
typedef float   v4f   __attribute__((ext_vector_type(4)));
typedef unsigned short v8us __attribute__((ext_vector_type(8)));
typedef unsigned short v4us __attribute__((ext_vector_type(4)));

union Frag16 {
    v16bf bf;
    v8us  h[2];
    unsigned short us[16];
};

__device__ __forceinline__ unsigned short f2bf(float f) {
    union { float f; unsigned int u; } x;
    x.f = f;
    unsigned int u = x.u;
    u += 0x7FFFu + ((u >> 16) & 1u);   // round-to-nearest-even
    return (unsigned short)(u >> 16);
}

__device__ __forceinline__ v16bf load_frag(const unsigned short* p0,
                                           const unsigned short* p1) {
    Frag16 f;
    f.h[0] = *(const v8us*)p0;   // global_load_b128
    f.h[1] = *(const v8us*)p1;   // global_load_b128
    return f.bf;
}

__device__ __forceinline__ v8f wmma_bf16(v16bf a, v16bf b, v8f c) {
    // v_wmma_f32_16x16x32_bf16
    return __builtin_amdgcn_wmma_f32_16x16x32_bf16(
        false, a, false, b, (short)0, c, false, false);
}

// ---------------------------------------------------------------------------
// fp32 -> bf16 conversion, 4 elems/thread
// ---------------------------------------------------------------------------
__global__ __launch_bounds__(256)
void f32_to_bf16_v4(const float* __restrict__ in,
                    unsigned short* __restrict__ out, int n) {
    int i = (blockIdx.x * blockDim.x + threadIdx.x) * 4;
    if (i >= n) return;
    v4f f = *(const v4f*)(in + i);
    union { v4us v; unsigned short s[4]; } u;
#pragma unroll
    for (int j = 0; j < 4; ++j) u.s[j] = f2bf(f[j]);
    *(v4us*)(out + i) = u.v;
}

// ---------------------------------------------------------------------------
// QKV GEMM: y[m,f] = sum_k x[m,k]*W[f,k] + b[f],  M=4096, N=3072, K=1024.
// Wave tile: 32(M) x 64(N) = 8 accumulators. WG = 8 waves = 64(M) x 256(N).
// Per k-step: 2 A frags + 4 B frags = 12 b128 loads feeding 8 WMMAs.
// Epilogue scatters into Q,K as [B,H,T,D] bf16 and V transposed as [B,H,D,T].
// ---------------------------------------------------------------------------
__global__ __launch_bounds__(256)
void qkv_gemm(const unsigned short* __restrict__ xbf,
              const unsigned short* __restrict__ wbf,
              const float* __restrict__ bias,
              unsigned short* __restrict__ qb,
              unsigned short* __restrict__ kb,
              unsigned short* __restrict__ vT) {
    const int K = 1024, T = 2048;
    int tid  = threadIdx.x;
    int wave = tid >> 5, lane = tid & 31;
    int r = lane & 15, half = lane >> 4;
    int m0 = blockIdx.x * 64  + (wave >> 2) * 32;
    int n0 = blockIdx.y * 256 + (wave & 3) * 64;

    v8f acc[2][4] = {};
    const unsigned short* arow0 = xbf + (size_t)(m0 + r)      * K;
    const unsigned short* arow1 = xbf + (size_t)(m0 + 16 + r) * K;

    for (int k = 0; k < K; k += 32) {
        v16bf a0 = load_frag(arow0 + k + half * 8, arow0 + k + 16 + half * 8);
        v16bf a1 = load_frag(arow1 + k + half * 8, arow1 + k + 16 + half * 8);
#pragma unroll
        for (int j = 0; j < 4; ++j) {
            const unsigned short* brow = wbf + (size_t)(n0 + j * 16 + r) * K;
            v16bf bf = load_frag(brow + k + half * 8, brow + k + 16 + half * 8);
            acc[0][j] = wmma_bf16(a0, bf, acc[0][j]);
            acc[1][j] = wmma_bf16(a1, bf, acc[1][j]);
        }
    }

#pragma unroll
    for (int mi = 0; mi < 2; ++mi) {
#pragma unroll
        for (int j = 0; j < 4; ++j) {
            int f   = n0 + j * 16 + r;     // output feature (uniform sec/tile)
            float bv = bias[f];
            int sec = f >> 10;             // 0=Q 1=K 2=V
            int idx = f & 1023;
            int hh  = idx >> 6, d = idx & 63;
#pragma unroll
            for (int v = 0; v < 8; ++v) {
                int m = m0 + mi * 16 + v + half * 8;     // token row
                int b = m >> 11, t = m & 2047;
                unsigned short val = f2bf(acc[mi][j][v] + bv);
                size_t hb = (size_t)(b * 16 + hh);
                if (sec == 0)      qb[(hb * T + t) * 64 + d] = val;
                else if (sec == 1) kb[(hb * T + t) * 64 + d] = val;
                else               vT[(hb * 64 + d) * T + t] = val;
            }
        }
    }
}

// ---------------------------------------------------------------------------
// Flash attention (causal). One wave per 16 query rows; 8 waves/WG -> 128 rows.
// Computes S^T = K.Q^T so each lane owns a full query row of scores; exp'd
// P^T lands directly in the B-fragment layout for O^T = V^T . P^T (no LDS).
// ---------------------------------------------------------------------------
__global__ __launch_bounds__(256)
void flash_attn(const unsigned short* __restrict__ qb,
                const unsigned short* __restrict__ kb,
                const unsigned short* __restrict__ vT,
                unsigned short* __restrict__ ybf) {
    const int T = 2048, D = 64;
    int tid  = threadIdx.x;
    int wave = tid >> 5, lane = tid & 31;
    int r = lane & 15, half = lane >> 4;
    int h = blockIdx.y, b = blockIdx.z;
    int q0 = blockIdx.x * 128 + wave * 16;

    size_t headoff = (size_t)(b * 16 + h) * T * D;
    const unsigned short* Qh = qb + headoff;
    const unsigned short* Kh = kb + headoff;
    const unsigned short* Vh = vT + headoff;      // [D, T]

    // Q B-fragments (cols = 16 query rows), reused across all key tiles
    const unsigned short* qrow = Qh + (size_t)(q0 + r) * D;
    v16bf bq0 = load_frag(qrow +      half * 8, qrow + 16 + half * 8);
    v16bf bq1 = load_frag(qrow + 32 + half * 8, qrow + 48 + half * 8);

    const float scale = 0.125f;                    // 1/sqrt(64)
    const float NEG_INF = -__builtin_inff();
    float mrun = NEG_INF, lrun = 0.0f;
    v8f oacc[4] = {};
    int q_abs  = q0 + r;
    int ntiles = (q0 + 16 + 31) >> 5;

    for (int kt = 0; kt < ntiles; ++kt) {
        int kbase = kt * 32;
        const unsigned short* k0row = Kh + (size_t)(kbase +      r) * D;
        const unsigned short* k1row = Kh + (size_t)(kbase + 16 + r) * D;
        v16bf a00 = load_frag(k0row +      half * 8, k0row + 16 + half * 8);
        v16bf a01 = load_frag(k0row + 32 + half * 8, k0row + 48 + half * 8);
        v16bf a10 = load_frag(k1row +      half * 8, k1row + 16 + half * 8);
        v16bf a11 = load_frag(k1row + 32 + half * 8, k1row + 48 + half * 8);

        v8f s0 = {}, s1 = {};
        s0 = wmma_bf16(a00, bq0, s0);
        s0 = wmma_bf16(a01, bq1, s0);
        s1 = wmma_bf16(a10, bq0, s1);
        s1 = wmma_bf16(a11, bq1, s1);

        // scale + causal mask; lane holds query row (q0+r), keys in-lane
        float sv[16];
        float tmax = NEG_INF;
#pragma unroll
        for (int j = 0; j < 8; ++j) {
            int key0 = kbase + j + half * 8;
            int key1 = kbase + 16 + j + half * 8;
            float v0 = (key0 <= q_abs) ? s0[j] * scale : NEG_INF;
            float v1 = (key1 <= q_abs) ? s1[j] * scale : NEG_INF;
            sv[j] = v0; sv[8 + j] = v1;
            tmax = fmaxf(tmax, fmaxf(v0, v1));
        }
        tmax = fmaxf(tmax, __shfl_xor(tmax, 16, 32));   // other key half
        float mnew  = fmaxf(mrun, tmax);
        float alpha = __expf(mrun - mnew);

        float psum = 0.0f;
        Frag16 pf;
#pragma unroll
        for (int i = 0; i < 16; ++i) {
            float p = __expf(sv[i] - mnew);
            psum += p;
            pf.us[i] = f2bf(p);          // exactly the PV B-fragment layout
        }
        psum += __shfl_xor(psum, 16, 32);
        lrun  = lrun * alpha + psum;
        mrun  = mnew;

#pragma unroll
        for (int dt = 0; dt < 4; ++dt)
#pragma unroll
            for (int j = 0; j < 8; ++j) oacc[dt][j] *= alpha;

#pragma unroll
        for (int dt = 0; dt < 4; ++dt) {
            const unsigned short* vrow = Vh + (size_t)(dt * 16 + r) * T + kbase;
            v16bf va = load_frag(vrow + half * 8, vrow + 16 + half * 8);
            oacc[dt] = wmma_bf16(va, pf.bf, oacc[dt]);
        }
    }

    float inv = 1.0f / lrun;
    int tok = b * T + q0 + r;
#pragma unroll
    for (int dt = 0; dt < 4; ++dt)
#pragma unroll
        for (int j = 0; j < 8; ++j) {
            int c = h * 64 + dt * 16 + j + half * 8;
            ybf[(size_t)tok * 1024 + c] = f2bf(oacc[dt][j] * inv);
        }
}

// ---------------------------------------------------------------------------
// Output projection: out[m,f] = sum_k y[m,k]*Wp[f,k] + bp[f]. M=4096,N=K=1024.
// Same 32Mx64N wave tile as qkv_gemm; fp32 epilogue into d_out.
// ---------------------------------------------------------------------------
__global__ __launch_bounds__(256)
void proj_gemm(const unsigned short* __restrict__ ybf,
               const unsigned short* __restrict__ wbf,
               const float* __restrict__ bias,
               float* __restrict__ out) {
    const int K = 1024;
    int tid  = threadIdx.x;
    int wave = tid >> 5, lane = tid & 31;
    int r = lane & 15, half = lane >> 4;
    int m0 = blockIdx.x * 64  + (wave >> 2) * 32;
    int n0 = blockIdx.y * 256 + (wave & 3) * 64;

    v8f acc[2][4] = {};
    const unsigned short* arow0 = ybf + (size_t)(m0 + r)      * K;
    const unsigned short* arow1 = ybf + (size_t)(m0 + 16 + r) * K;

    for (int k = 0; k < K; k += 32) {
        v16bf a0 = load_frag(arow0 + k + half * 8, arow0 + k + 16 + half * 8);
        v16bf a1 = load_frag(arow1 + k + half * 8, arow1 + k + 16 + half * 8);
#pragma unroll
        for (int j = 0; j < 4; ++j) {
            const unsigned short* brow = wbf + (size_t)(n0 + j * 16 + r) * K;
            v16bf bf = load_frag(brow + k + half * 8, brow + k + 16 + half * 8);
            acc[0][j] = wmma_bf16(a0, bf, acc[0][j]);
            acc[1][j] = wmma_bf16(a1, bf, acc[1][j]);
        }
    }

#pragma unroll
    for (int mi = 0; mi < 2; ++mi) {
#pragma unroll
        for (int j = 0; j < 4; ++j) {
            int f = n0 + j * 16 + r;
            float bv = bias[f];
#pragma unroll
            for (int v = 0; v < 8; ++v) {
                int m = m0 + mi * 16 + v + half * 8;
                out[(size_t)m * 1024 + f] = acc[mi][j][v] + bv;
            }
        }
    }
}

// ---------------------------------------------------------------------------
// Launch
// ---------------------------------------------------------------------------
extern "C" void kernel_launch(void* const* d_in, const int* in_sizes, int n_in,
                              void* d_out, int out_size, void* d_ws, size_t ws_size,
                              hipStream_t stream) {
    const float* x      = (const float*)d_in[0];
    const float* W_attn = (const float*)d_in[1];
    const float* b_attn = (const float*)d_in[2];
    const float* W_proj = (const float*)d_in[3];
    const float* b_proj = (const float*)d_in[4];
    float* out = (float*)d_out;

    // Workspace carve-up (bf16 elements), total ~50.3 MB -> L2-resident
    unsigned short* xbf   = (unsigned short*)d_ws;
    unsigned short* wattn = xbf   + 4194304;   // 4096*1024
    unsigned short* wproj = wattn + 3145728;   // 3072*1024
    unsigned short* qb    = wproj + 1048576;   // 1024*1024
    unsigned short* kb    = qb    + 4194304;   // [B,H,T,D]
    unsigned short* vT    = kb    + 4194304;   // [B,H,T,D]
    unsigned short* ybf   = vT    + 4194304;   // [B,H,D,T]

    f32_to_bf16_v4<<<4096, 256, 0, stream>>>(x,      xbf,   4194304);
    f32_to_bf16_v4<<<3072, 256, 0, stream>>>(W_attn, wattn, 3145728);
    f32_to_bf16_v4<<<1024, 256, 0, stream>>>(W_proj, wproj, 1048576);

    qkv_gemm <<<dim3(64, 12),    256, 0, stream>>>(xbf, wattn, b_attn, qb, kb, vT);
    flash_attn<<<dim3(16, 16, 2), 256, 0, stream>>>(qb, kb, vT, ybf);
    proj_gemm<<<dim3(64, 4),     256, 0, stream>>>(ybf, wproj, b_proj, out);
}
